// GGNNMeanMixedResidual_78151224918836
// MI455X (gfx1250) — matrix operand order repaired and checked by reading, not compiled, not run on hardware
//
#include <hip/hip_runtime.h>
#include <cstdint>

#define N_NODES 50000
#define N_EDGES 300000
#define N_ETYPES 4
#define DD      256
#define FEAT    1024   /* D + RES */
#define N_GRAPHS 128
#define N_STEPS  8
#define HID      256

typedef __attribute__((ext_vector_type(16))) __bf16 v16bf;
typedef __attribute__((ext_vector_type(8)))  float  v8f;

union Frag16 { v16bf v; uint4 u[2]; };

__device__ __forceinline__ unsigned short f2bf(float f) {
  unsigned u = __float_as_uint(f);
  u = u + 0x7FFFu + ((u >> 16) & 1u);   // round-to-nearest-even
  return (unsigned short)(u >> 16);
}

// ---------------------------------------------------------------- utilities
__global__ void zero_f32(float* p, int n) {
  int i = blockIdx.x * blockDim.x + threadIdx.x;
  if (i < n) p[i] = 0.0f;
}
__global__ void zero_i32(int* p, int n) {
  int i = blockIdx.x * blockDim.x + threadIdx.x;
  if (i < n) p[i] = 0;
}
__global__ void f32_to_bf16_k(const float* __restrict__ in, unsigned short* __restrict__ out, int n) {
  int i = blockIdx.x * blockDim.x + threadIdx.x;
  if (i < n) out[i] = f2bf(in[i]);
}

// Pack a row-major f32 KxN matrix into WMMA B-fragment order (bf16).
// Layout: Bp[((kt*(N/16) + nt)*32 + lane)*16 + j] = B[kt*32 + (lane>>4)*8 + (j&7) + (j>>3)*16][nt*16 + (lane&15)]
__global__ void pack_b_kernel(const float* __restrict__ B, unsigned short* __restrict__ Bp, int K, int N) {
  int idx = blockIdx.x * blockDim.x + threadIdx.x;
  if (idx >= K * N) return;
  int j    = idx & 15;
  int lane = (idx >> 4) & 31;
  int tile = idx >> 9;
  int ntiles = N >> 4;
  int kt = tile / ntiles;
  int nt = tile - kt * ntiles;
  int k = kt * 32 + (lane >> 4) * 8 + (j & 7) + (j >> 3) * 16;
  int n = nt * 16 + (lane & 15);
  Bp[idx] = f2bf(B[(size_t)k * N + n]);
}

// ------------------------------------------------------------ bf16 WMMA GEMM
// C[M,N] (f32) = A[M,K] (bf16 row-major) * Bpacked + bias (optional).
// One wave computes a 16(M) x 64(N) tile; K-loop in chunks of 32.
__global__ __launch_bounds__(256)
void gemm_bf16(const unsigned short* __restrict__ A,
               const unsigned short* __restrict__ Bp,
               const float* __restrict__ bias,
               float* __restrict__ C,
               int M, int N, int K)
{
  int wave = (int)((blockIdx.x * blockDim.x + threadIdx.x) >> 5);
  int lane = threadIdx.x & 31;
  int nGroups = N >> 6;                 // 64-wide groups
  int mTiles  = (M + 15) >> 4;
  if (wave >= mTiles * nGroups) return;
  int mTile  = wave / nGroups;
  int nGroup = wave - mTile * nGroups;
  int hl  = lane >> 4;                  // half of wave (K interleave)
  int row = mTile * 16 + (lane & 15);   // A row for this lane (M = lane%16)
  const unsigned short* Arow = A + (size_t)row * K;
  int nTileBase = nGroup * 4;           // in 16-wide tiles
  int ntiles16  = N >> 4;

  v8f acc0 = {}, acc1 = {}, acc2 = {}, acc3 = {};
  int ktiles = K >> 5;
  for (int kt = 0; kt < ktiles; ++kt) {
    Frag16 af;
    int k0 = kt * 32 + hl * 8;
    af.u[0] = *(const uint4*)(Arow + k0);        // elems j=0..7  : K = kt*32 + hl*8 + j
    af.u[1] = *(const uint4*)(Arow + k0 + 16);   // elems j=8..15 : K = +16
    const unsigned short* bbase =
        Bp + ((((size_t)kt * ntiles16) + nTileBase) * 32 + lane) * 16;
    Frag16 b0, b1, b2, b3;
    b0.u[0] = *(const uint4*)(bbase);            b0.u[1] = *(const uint4*)(bbase + 8);
    b1.u[0] = *(const uint4*)(bbase + 32*16);    b1.u[1] = *(const uint4*)(bbase + 32*16 + 8);
    b2.u[0] = *(const uint4*)(bbase + 2*32*16);  b2.u[1] = *(const uint4*)(bbase + 2*32*16 + 8);
    b3.u[0] = *(const uint4*)(bbase + 3*32*16);  b3.u[1] = *(const uint4*)(bbase + 3*32*16 + 8);
    acc0 = __builtin_amdgcn_wmma_f32_16x16x32_bf16(false, af.v, false, b0.v, (short)0, acc0, false, false);
    acc1 = __builtin_amdgcn_wmma_f32_16x16x32_bf16(false, af.v, false, b1.v, (short)0, acc1, false, false);
    acc2 = __builtin_amdgcn_wmma_f32_16x16x32_bf16(false, af.v, false, b2.v, (short)0, acc2, false, false);
    acc3 = __builtin_amdgcn_wmma_f32_16x16x32_bf16(false, af.v, false, b3.v, (short)0, acc3, false, false);
  }

  v8f accs[4] = {acc0, acc1, acc2, acc3};
  #pragma unroll
  for (int j = 0; j < 4; ++j) {
    int col = (nTileBase + j) * 16 + (lane & 15);
    float bv = bias ? bias[col] : 0.0f;
    #pragma unroll
    for (int v = 0; v < 8; ++v) {
      int r = mTile * 16 + v + hl * 8;           // C layout: VGPR v -> M = v + 8*(lane/16)
      C[(size_t)r * N + col] = accs[j][v] + bv;
    }
  }
}

// ------------------------------------------------------------ graph kernels
__global__ void init_h_kernel(const float* __restrict__ features,
                              float* __restrict__ h, unsigned short* __restrict__ h_bf) {
  int idx = blockIdx.x * blockDim.x + threadIdx.x;
  if (idx >= N_NODES * DD) return;
  int r = idx >> 8, c = idx & 255;
  float v = features[(size_t)r * FEAT + c];
  h[idx] = v;
  h_bf[idx] = f2bf(v);
}

__global__ void deg_count_kernel(const int* __restrict__ dst, const int* __restrict__ et,
                                 int* __restrict__ deg) {
  int e = blockIdx.x * blockDim.x + threadIdx.x;
  if (e >= N_EDGES) return;
  atomicAdd(&deg[dst[e] * N_ETYPES + et[e]], 1);
}

// a[m,c] = sum_t deg[m,t] * b_e[t,c]   (exact per-message bias contribution)
__global__ void a_init_kernel(const int* __restrict__ deg, const float* __restrict__ b_e,
                              float* __restrict__ a) {
  int idx = blockIdx.x * blockDim.x + threadIdx.x;
  if (idx >= N_NODES * DD) return;
  int r = idx >> 8, c = idx & 255;
  const int* d = deg + (size_t)r * N_ETYPES;
  a[idx] = (float)d[0] * b_e[c] + (float)d[1] * b_e[DD + c]
         + (float)d[2] * b_e[2 * DD + c] + (float)d[3] * b_e[3 * DD + c];
}

// One wave per edge; adds ht[src,:] into a[dst,:] for edges of type t.
__global__ __launch_bounds__(256)
void edge_scatter_kernel(const float* __restrict__ ht,
                         const int* __restrict__ src, const int* __restrict__ dst,
                         const int* __restrict__ et, int t, float* __restrict__ a) {
  int e = (int)((blockIdx.x * blockDim.x + threadIdx.x) >> 5);
  int lane = threadIdx.x & 31;
  if (e >= N_EDGES) return;
  if (et[e] != t) return;
  const float* m = ht + (size_t)src[e] * DD + lane * 8;
  float*       o = a  + (size_t)dst[e] * DD + lane * 8;
  float4 v0 = *(const float4*)m;
  float4 v1 = *(const float4*)(m + 4);
  unsafeAtomicAdd(o + 0, v0.x); unsafeAtomicAdd(o + 1, v0.y);
  unsafeAtomicAdd(o + 2, v0.z); unsafeAtomicAdd(o + 3, v0.w);
  unsafeAtomicAdd(o + 4, v1.x); unsafeAtomicAdd(o + 5, v1.y);
  unsafeAtomicAdd(o + 6, v1.z); unsafeAtomicAdd(o + 7, v1.w);
}

__global__ void gru_cell_kernel(const float* __restrict__ gi, const float* __restrict__ gh,
                                float* __restrict__ h, unsigned short* __restrict__ h_bf) {
  int idx = blockIdx.x * blockDim.x + threadIdx.x;
  if (idx >= N_NODES * DD) return;
  int r = idx >> 8, c = idx & 255;
  size_t base = (size_t)r * (3 * DD);
  float ir = gi[base + c], iz = gi[base + DD + c], in_ = gi[base + 2 * DD + c];
  float hr = gh[base + c], hz = gh[base + DD + c], hn  = gh[base + 2 * DD + c];
  float rg = 1.0f / (1.0f + expf(-(ir + hr)));
  float zg = 1.0f / (1.0f + expf(-(iz + hz)));
  float nn = tanhf(in_ + rg * hn);
  float hv = h[idx];
  float o  = (1.0f - zg) * nn + zg * hv;
  h[idx] = o;
  h_bf[idx] = f2bf(o);
}

__global__ void pool_cnt_kernel(const int* __restrict__ n2g, float* __restrict__ cnt) {
  int r = blockIdx.x * blockDim.x + threadIdx.x;
  if (r >= N_NODES) return;
  unsafeAtomicAdd(&cnt[n2g[r]], 1.0f);
}

__global__ void pool_accum_kernel(const float* __restrict__ h, const float* __restrict__ features,
                                  const int* __restrict__ n2g, float* __restrict__ pooled) {
  long long idx = (long long)blockIdx.x * blockDim.x + threadIdx.x;
  if (idx >= (long long)N_NODES * FEAT) return;
  int r = (int)(idx >> 10);
  int c = (int)(idx & 1023);
  float v = (c < DD) ? h[(size_t)r * DD + c] : features[(size_t)r * FEAT + c];
  unsafeAtomicAdd(&pooled[(size_t)n2g[r] * FEAT + c], v);
}

__global__ __launch_bounds__(256)
void mlp_kernel(const float* __restrict__ pooled, const float* __restrict__ cnt,
                const float* __restrict__ W1, const float* __restrict__ b1,
                const float* __restrict__ W2, const float* __restrict__ b2,
                float* __restrict__ out) {
  int g = blockIdx.x;
  int j = threadIdx.x;          // HID = 256 threads
  float inv = 1.0f / fmaxf(cnt[g], 1.0f);
  const float* p = pooled + (size_t)g * FEAT;
  float acc = b1[j];
  for (int c = 0; c < FEAT; ++c) acc += p[c] * inv * W1[(size_t)c * HID + j];
  acc = fmaxf(acc, 0.0f) * W2[j];
  __shared__ float red[HID];
  red[j] = acc;
  __syncthreads();
  for (int s = HID / 2; s > 0; s >>= 1) {
    if (j < s) red[j] += red[j + s];
    __syncthreads();
  }
  if (j == 0) out[g] = 1.0f / (1.0f + expf(-(red[0] + b2[0])));
}

// ---------------------------------------------------------------- launcher
extern "C" void kernel_launch(void* const* d_in, const int* in_sizes, int n_in,
                              void* d_out, int out_size, void* d_ws, size_t ws_size,
                              hipStream_t stream) {
  const float* features = (const float*)d_in[0];
  const float* W_e  = (const float*)d_in[1];
  const float* b_e  = (const float*)d_in[2];
  const float* W_ih = (const float*)d_in[3];
  const float* W_hh = (const float*)d_in[4];
  const float* b_ih = (const float*)d_in[5];
  const float* b_hh = (const float*)d_in[6];
  const float* W1   = (const float*)d_in[7];
  const float* b1   = (const float*)d_in[8];
  const float* W2   = (const float*)d_in[9];
  const float* b2   = (const float*)d_in[10];
  const int* src = (const int*)d_in[11];
  const int* dst = (const int*)d_in[12];
  const int* ety = (const int*)d_in[13];
  const int* n2g = (const int*)d_in[14];
  float* out = (float*)d_out;
  (void)in_sizes; (void)n_in; (void)out_size; (void)ws_size;

  char* base = (char*)d_ws;
  size_t off = 0;
  auto take = [&](size_t bytes) -> char* {
    char* p = base + off;
    off += (bytes + 255) & ~(size_t)255;
    return p;
  };
  float* h   = (float*)take((size_t)N_NODES * DD * 4);
  float* a   = (float*)take((size_t)N_NODES * DD * 4);
  float* ht  = (float*)take((size_t)N_NODES * DD * 4);
  float* gi  = (float*)take((size_t)N_NODES * 3 * DD * 4);
  float* gh  = (float*)take((size_t)N_NODES * 3 * DD * 4);
  unsigned short* h_bf = (unsigned short*)take((size_t)N_NODES * DD * 2);
  unsigned short* a_bf = (unsigned short*)take((size_t)N_NODES * DD * 2);
  unsigned short* weP  = (unsigned short*)take((size_t)N_ETYPES * DD * DD * 2);
  unsigned short* wihP = (unsigned short*)take((size_t)DD * 3 * DD * 2);
  unsigned short* whhP = (unsigned short*)take((size_t)DD * 3 * DD * 2);
  int*   deg    = (int*)take((size_t)N_NODES * N_ETYPES * 4);
  float* pooled = (float*)take((size_t)N_GRAPHS * FEAT * 4);
  float* cnt    = (float*)take((size_t)N_GRAPHS * 4);

  const int TB = 256;
  auto cdiv = [](long long x, long long y) { return (int)((x + y - 1) / y); };

  // --- one-time (per launch) weight packing + in-degree counts
  for (int t = 0; t < N_ETYPES; ++t)
    pack_b_kernel<<<cdiv(DD * DD, TB), TB, 0, stream>>>(
        W_e + (size_t)t * DD * DD, weP + (size_t)t * DD * DD, DD, DD);
  pack_b_kernel<<<cdiv(DD * 3 * DD, TB), TB, 0, stream>>>(W_ih, wihP, DD, 3 * DD);
  pack_b_kernel<<<cdiv(DD * 3 * DD, TB), TB, 0, stream>>>(W_hh, whhP, DD, 3 * DD);

  zero_i32<<<cdiv(N_NODES * N_ETYPES, TB), TB, 0, stream>>>(deg, N_NODES * N_ETYPES);
  deg_count_kernel<<<cdiv(N_EDGES, TB), TB, 0, stream>>>(dst, ety, deg);

  init_h_kernel<<<cdiv((long long)N_NODES * DD, TB), TB, 0, stream>>>(features, h, h_bf);

  // --- message passing steps
  const int wavesE = (N_NODES / 16) * (DD / 64);       // etype GEMM waves
  const int wavesG = (N_NODES / 16) * ((3 * DD) / 64); // gate GEMM waves
  for (int s = 0; s < N_STEPS; ++s) {
    a_init_kernel<<<cdiv((long long)N_NODES * DD, TB), TB, 0, stream>>>(deg, b_e, a);
    for (int t = 0; t < N_ETYPES; ++t) {
      gemm_bf16<<<cdiv((long long)wavesE * 32, TB), TB, 0, stream>>>(
          h_bf, weP + (size_t)t * DD * DD, nullptr, ht, N_NODES, DD, DD);
      edge_scatter_kernel<<<cdiv((long long)N_EDGES * 32, TB), TB, 0, stream>>>(
          ht, src, dst, ety, t, a);
    }
    f32_to_bf16_k<<<cdiv((long long)N_NODES * DD, TB), TB, 0, stream>>>(a, a_bf, N_NODES * DD);
    gemm_bf16<<<cdiv((long long)wavesG * 32, TB), TB, 0, stream>>>(
        a_bf, wihP, b_ih, gi, N_NODES, 3 * DD, DD);
    gemm_bf16<<<cdiv((long long)wavesG * 32, TB), TB, 0, stream>>>(
        h_bf, whhP, b_hh, gh, N_NODES, 3 * DD, DD);
    gru_cell_kernel<<<cdiv((long long)N_NODES * DD, TB), TB, 0, stream>>>(gi, gh, h, h_bf);
  }

  // --- readout
  zero_f32<<<cdiv(N_GRAPHS * FEAT, TB), TB, 0, stream>>>(pooled, N_GRAPHS * FEAT);
  zero_f32<<<1, TB, 0, stream>>>(cnt, N_GRAPHS);
  pool_cnt_kernel<<<cdiv(N_NODES, TB), TB, 0, stream>>>(n2g, cnt);
  pool_accum_kernel<<<cdiv((long long)N_NODES * FEAT, TB), TB, 0, stream>>>(h, features, n2g, pooled);
  mlp_kernel<<<N_GRAPHS, HID, 0, stream>>>(pooled, cnt, W1, b1, W2, b2, out);
}
